// StandaloneCGCNN_8624294330726
// MI455X (gfx1250) — matrix operand order repaired and checked by reading, not compiled, not run on hardware
//
#include <hip/hip_runtime.h>
#include <hip/hip_bf16.h>
#include <math.h>

#define HID 128
#define TPB 128  // 4 waves of 32

typedef __bf16 bf16_t;
typedef __attribute__((ext_vector_type(16))) __bf16 bf16x16;
typedef __attribute__((ext_vector_type(8)))  float  f32x8;

union FragU { bf16x16 v; uint4 q[2]; };

__device__ __forceinline__ f32x8 wmma_bf16(bf16x16 a, bf16x16 b, f32x8 c) {
  // D(f32 16x16) = A(16x32 bf16) * B(32x16 bf16) + C
  return __builtin_amdgcn_wmma_f32_16x16x32_bf16(false, a, false, b,
                                                 (short)0, c, false, false);
}

// A fragment (16x32 bf16): lane holds row = lane&15.
// lanes 0-15 hold K={0..7,16..23}; lanes 16-31 hold K={8..15,24..31}.
__device__ __forceinline__ bf16x16 load_a_frag(const bf16_t* rowp, int lane) {
  const int half = (lane < 16) ? 0 : 8;
  FragU f;
  f.q[0] = *reinterpret_cast<const uint4*>(rowp + half);
  f.q[1] = *reinterpret_cast<const uint4*>(rowp + half + 16);
  return f.v;
}

// B fragment (32x16 bf16) from pre-packed weights: lane-major, 16 contiguous
// bf16 per lane (=32B) -> two 16B loads.
__device__ __forceinline__ bf16x16 load_b_frag(const bf16_t* packed, int frag, int lane) {
  const bf16_t* p = packed + (((size_t)frag * 32 + lane) << 4);
  FragU f;
  f.q[0] = *reinterpret_cast<const uint4*>(p);
  f.q[1] = *reinterpret_cast<const uint4*>(p + 8);
  return f.v;
}

// ---------------- helper kernels ----------------

__global__ void k_zerof(float* __restrict__ p, int n) {
  int t = blockIdx.x * blockDim.x + threadIdx.x;
  if (t < n) p[t] = 0.f;
}

__global__ void k_embed(const int* __restrict__ types, const float* __restrict__ emb,
                        float* __restrict__ x, int nAtoms) {
  int t = blockIdx.x * blockDim.x + threadIdx.x;
  if (t >= nAtoms * HID) return;
  int i = t >> 7, h = t & 127;
  x[t] = emb[types[i] * HID + h];
}

__global__ void k_cast_bf16(const float* __restrict__ in, bf16_t* __restrict__ out, int n) {
  int t = blockIdx.x * blockDim.x + threadIdx.x;
  if (t < n) out[t] = (bf16_t)in[t];
}

// Repack a row-major f32 weight [K][128] into WMMA B-fragment order (bf16).
// frag = kt*8+nt ; within frag: lane L (32), elem e (16):
//   k = kt*32 + (L>=16 ? 8 : 0) + (e&7) + (e>=8 ? 16 : 0) ; n = nt*16 + (L&15)
__global__ void k_pack(const float* __restrict__ W, bf16_t* __restrict__ P, int K) {
  int t = blockIdx.x * blockDim.x + threadIdx.x;
  if (t >= K * HID) return;
  int e    = t & 15;
  int lane = (t >> 4) & 31;
  int frag = t >> 9;
  int kt = frag >> 3, nt = frag & 7;
  int k = kt * 32 + ((lane >= 16) ? 8 : 0) + (e & 7) + ((e & 8) ? 16 : 0);
  int n = nt * 16 + (lane & 15);
  P[t] = (bf16_t)W[k * HID + n];
}

// ---------------- xt = x @ Wn + bn (bf16 out), 32 rows/wave ----------------

__global__ __launch_bounds__(TPB) void k_node_linear(
    const bf16_t* __restrict__ x_bf, const bf16_t* __restrict__ Wp,
    const float* __restrict__ bias, bf16_t* __restrict__ out_bf, int nRows) {
  const int lane = threadIdx.x & 31, wave = threadIdx.x >> 5;
  const int rbase = blockIdx.x * 128 + wave * 32;
  const int r = lane & 15;
  const int nl = lane & 15;
  const int rowoff = (lane < 16) ? 0 : 8;
  const int crow0 = min(rbase + r, nRows - 1);
  const int crow1 = min(rbase + 16 + r, nRows - 1);

  f32x8 acc[2][8];
#pragma unroll
  for (int nt = 0; nt < 8; ++nt) {
    float bv = bias[nt * 16 + nl];
    acc[0][nt] = f32x8{bv, bv, bv, bv, bv, bv, bv, bv};
    acc[1][nt] = acc[0][nt];
  }
#pragma unroll
  for (int kt = 0; kt < 4; ++kt) {
    bf16x16 a0 = load_a_frag(x_bf + (size_t)crow0 * HID + kt * 32, lane);
    bf16x16 a1 = load_a_frag(x_bf + (size_t)crow1 * HID + kt * 32, lane);
#pragma unroll
    for (int nt = 0; nt < 8; ++nt) {
      bf16x16 b = load_b_frag(Wp, kt * 8 + nt, lane);
      acc[0][nt] = wmma_bf16(a0, b, acc[0][nt]);
      acc[1][nt] = wmma_bf16(a1, b, acc[1][nt]);
    }
  }
#pragma unroll
  for (int t = 0; t < 2; ++t) {
#pragma unroll
    for (int nt = 0; nt < 8; ++nt) {
      const int col = nt * 16 + nl;
#pragma unroll
      for (int v = 0; v < 8; ++v) {
        const int rr = rbase + t * 16 + rowoff + v;
        if (rr < nRows) out_bf[(size_t)rr * HID + col] = (bf16_t)acc[t][nt][v];
      }
    }
  }
}

// ---------------- edge message MLP + scatter-add, 32 edges/wave ----------------
// msg = (relu([xt[dst],xt[src]]@Wm1 + bm1) @ Wm2 + bm2) * (d*We + be)

__global__ __launch_bounds__(TPB) void k_edge_message(
    const bf16_t* __restrict__ xt_bf,
    const bf16_t* __restrict__ Wm1p, const bf16_t* __restrict__ Wm2p,
    const float* __restrict__ bm1, const float* __restrict__ bm2,
    const float* __restrict__ We, const float* __restrict__ be,
    const int* __restrict__ src, const int* __restrict__ dst,
    const float* __restrict__ dist, float* __restrict__ agg, int nEdges) {
  __shared__ int   s_dst[128];
  __shared__ int   s_src[128];
  __shared__ float s_d[128];
  __shared__ __align__(16) bf16_t s_hid[4][32 * HID];  // 32KB

  const int tid = threadIdx.x;
  const int lane = tid & 31, wave = tid >> 5;
  const int ebase = blockIdx.x * 128;

  {
    int e = ebase + tid;
    bool ok = (e < nEdges);
    s_dst[tid] = ok ? dst[e] : -1;
    s_src[tid] = ok ? src[e] : 0;
    s_d[tid]   = ok ? dist[e] : 0.f;
  }
  if (tid == 0 && ebase + 128 < nEdges) {
    __builtin_prefetch(dst + ebase + 128, 0, 1);
    __builtin_prefetch(src + ebase + 128, 0, 1);
    __builtin_prefetch(dist + ebase + 128, 0, 1);
  }
  __syncthreads();

  const int r = lane & 15;
  const int nl = lane & 15;
  const int rowoff = (lane < 16) ? 0 : 8;
  const int slot0 = wave * 32 + r;       // A-gather slots for the two 16-edge tiles
  const int slot1 = slot0 + 16;

  // ---- GEMM1: hidden[32,128] = A[32,256] @ Wm1, A = [xt[dst] | xt[src]]
  f32x8 acc[2][8];
#pragma unroll
  for (int nt = 0; nt < 8; ++nt) {
    float bv = bm1[nt * 16 + nl];
    acc[0][nt] = f32x8{bv, bv, bv, bv, bv, bv, bv, bv};
    acc[1][nt] = acc[0][nt];
  }
#pragma unroll
  for (int kt = 0; kt < 8; ++kt) {
    int i0 = (kt < 4) ? s_dst[slot0] : s_src[slot0];
    int i1 = (kt < 4) ? s_dst[slot1] : s_src[slot1];
    if (i0 < 0) i0 = 0;
    if (i1 < 0) i1 = 0;
    const int ko = (kt & 3) * 32;
    bf16x16 a0 = load_a_frag(xt_bf + (size_t)i0 * HID + ko, lane);
    bf16x16 a1 = load_a_frag(xt_bf + (size_t)i1 * HID + ko, lane);
#pragma unroll
    for (int nt = 0; nt < 8; ++nt) {
      bf16x16 b = load_b_frag(Wm1p, kt * 8 + nt, lane);
      acc[0][nt] = wmma_bf16(a0, b, acc[0][nt]);
      acc[1][nt] = wmma_bf16(a1, b, acc[1][nt]);
    }
  }
  // relu -> bf16, stage in per-wave LDS (C-layout -> row-major A source)
#pragma unroll
  for (int t = 0; t < 2; ++t)
#pragma unroll
    for (int nt = 0; nt < 8; ++nt) {
      int col = nt * 16 + nl;
#pragma unroll
      for (int v = 0; v < 8; ++v)
        s_hid[wave][(t * 16 + v + rowoff) * HID + col] =
            (bf16_t)fmaxf(acc[t][nt][v], 0.f);
    }

  // ---- GEMM2: msg[32,128] = hidden @ Wm2
  f32x8 accm[2][8];
#pragma unroll
  for (int nt = 0; nt < 8; ++nt) {
    float bv = bm2[nt * 16 + nl];
    accm[0][nt] = f32x8{bv, bv, bv, bv, bv, bv, bv, bv};
    accm[1][nt] = accm[0][nt];
  }
#pragma unroll
  for (int kt = 0; kt < 4; ++kt) {
    bf16x16 a0 = load_a_frag(&s_hid[wave][r * HID + kt * 32], lane);
    bf16x16 a1 = load_a_frag(&s_hid[wave][(16 + r) * HID + kt * 32], lane);
#pragma unroll
    for (int nt = 0; nt < 8; ++nt) {
      bf16x16 b = load_b_frag(Wm2p, kt * 8 + nt, lane);
      accm[0][nt] = wmma_bf16(a0, b, accm[0][nt]);
      accm[1][nt] = wmma_bf16(a1, b, accm[1][nt]);
    }
  }

  // ---- msg *= (d*We + be) ; agg[dst] += msg
#pragma unroll
  for (int nt = 0; nt < 8; ++nt) {
    const int col = nt * 16 + nl;
    const float wE = We[col], bE = be[col];
#pragma unroll
    for (int t = 0; t < 2; ++t)
#pragma unroll
      for (int v = 0; v < 8; ++v) {
        const int slot = wave * 32 + t * 16 + rowoff + v;
        const int didx = s_dst[slot];
        if (didx >= 0) {
          float et = s_d[slot] * wE + bE;
          atomicAdd(&agg[(size_t)didx * HID + col], accm[t][nt][v] * et);
        }
      }
  }
}

// ---------------- node update, 32 rows/wave, shuffle layernorm ----------------
// x_out = LN(softplus(relu([agg,x]@Wu1+bu1)@Wu2+bu2)) (+ x_old if residual)

__global__ __launch_bounds__(TPB) void k_node_update(
    const bf16_t* __restrict__ agg_bf, const bf16_t* __restrict__ x_bf,
    const float* __restrict__ x_old,
    const bf16_t* __restrict__ Wu1p, const bf16_t* __restrict__ Wu2p,
    const float* __restrict__ bu1, const float* __restrict__ bu2,
    const float* __restrict__ ln_g, const float* __restrict__ ln_b,
    float* __restrict__ x_out, int nRows, int residual) {
  __shared__ __align__(16) bf16_t s_hid[4][32 * HID];  // 32KB

  const int lane = threadIdx.x & 31, wave = threadIdx.x >> 5;
  const int rbase = blockIdx.x * 128 + wave * 32;
  const int r = lane & 15;
  const int nl = lane & 15;
  const int rowoff = (lane < 16) ? 0 : 8;
  const int crow0 = min(rbase + r, nRows - 1);
  const int crow1 = min(rbase + 16 + r, nRows - 1);

  // ---- GEMM1: A = [agg | x] (bf16), K=256
  f32x8 acc[2][8];
#pragma unroll
  for (int nt = 0; nt < 8; ++nt) {
    float bv = bu1[nt * 16 + nl];
    acc[0][nt] = f32x8{bv, bv, bv, bv, bv, bv, bv, bv};
    acc[1][nt] = acc[0][nt];
  }
#pragma unroll
  for (int kt = 0; kt < 8; ++kt) {
    const bf16_t* srcm = (kt < 4) ? agg_bf : x_bf;
    const int ko = (kt & 3) * 32;
    bf16x16 a0 = load_a_frag(srcm + (size_t)crow0 * HID + ko, lane);
    bf16x16 a1 = load_a_frag(srcm + (size_t)crow1 * HID + ko, lane);
#pragma unroll
    for (int nt = 0; nt < 8; ++nt) {
      bf16x16 b = load_b_frag(Wu1p, kt * 8 + nt, lane);
      acc[0][nt] = wmma_bf16(a0, b, acc[0][nt]);
      acc[1][nt] = wmma_bf16(a1, b, acc[1][nt]);
    }
  }
#pragma unroll
  for (int t = 0; t < 2; ++t)
#pragma unroll
    for (int nt = 0; nt < 8; ++nt) {
      int col = nt * 16 + nl;
#pragma unroll
      for (int v = 0; v < 8; ++v)
        s_hid[wave][(t * 16 + v + rowoff) * HID + col] =
            (bf16_t)fmaxf(acc[t][nt][v], 0.f);
    }

  // ---- GEMM2 (K=128) -> softplus (kept in registers, C layout)
  f32x8 acc2[2][8];
#pragma unroll
  for (int nt = 0; nt < 8; ++nt) {
    float bv = bu2[nt * 16 + nl];
    acc2[0][nt] = f32x8{bv, bv, bv, bv, bv, bv, bv, bv};
    acc2[1][nt] = acc2[0][nt];
  }
#pragma unroll
  for (int kt = 0; kt < 4; ++kt) {
    bf16x16 a0 = load_a_frag(&s_hid[wave][r * HID + kt * 32], lane);
    bf16x16 a1 = load_a_frag(&s_hid[wave][(16 + r) * HID + kt * 32], lane);
#pragma unroll
    for (int nt = 0; nt < 8; ++nt) {
      bf16x16 b = load_b_frag(Wu2p, kt * 8 + nt, lane);
      acc2[0][nt] = wmma_bf16(a0, b, acc2[0][nt]);
      acc2[1][nt] = wmma_bf16(a1, b, acc2[1][nt]);
    }
  }
#pragma unroll
  for (int t = 0; t < 2; ++t)
#pragma unroll
    for (int nt = 0; nt < 8; ++nt)
#pragma unroll
      for (int v = 0; v < 8; ++v) {
        float h = acc2[t][nt][v];
        acc2[t][nt][v] = (h > 20.f) ? h : log1pf(__expf(h));
      }

  // ---- layernorm stats via cross-lane xor reductions:
  // row M = t*16 + v + rowoff lives at fixed (t,v) across the 16-lane half.
  float mu[2][8], rs[2][8];
#pragma unroll
  for (int t = 0; t < 2; ++t)
#pragma unroll
    for (int v = 0; v < 8; ++v) {
      float s = 0.f, s2 = 0.f;
#pragma unroll
      for (int nt = 0; nt < 8; ++nt) {
        float xx = acc2[t][nt][v];
        s += xx;
        s2 += xx * xx;
      }
#pragma unroll
      for (int m = 1; m < 16; m <<= 1) {
        s  += __shfl_xor(s, m, 32);
        s2 += __shfl_xor(s2, m, 32);
      }
      float m1 = s * (1.f / HID);
      float var = s2 * (1.f / HID) - m1 * m1;
      mu[t][v] = m1;
      rs[t][v] = rsqrtf(var + 1e-5f);
    }

  // ---- normalize + affine + residual, direct store from C layout
#pragma unroll
  for (int nt = 0; nt < 8; ++nt) {
    const int col = nt * 16 + nl;
    const float g = ln_g[col], b = ln_b[col];
#pragma unroll
    for (int t = 0; t < 2; ++t)
#pragma unroll
      for (int v = 0; v < 8; ++v) {
        const int grow = rbase + t * 16 + rowoff + v;
        if (grow < nRows) {
          float val = (acc2[t][nt][v] - mu[t][v]) * rs[t][v] * g + b;
          if (residual) val += x_old[(size_t)grow * HID + col];
          x_out[(size_t)grow * HID + col] = val;
        }
      }
  }
}

// ---------------- pooling + head ----------------

__global__ void k_pool(const float* __restrict__ x, const int* __restrict__ bids,
                       float* __restrict__ sums, float* __restrict__ cnt, int nAtoms) {
  int t = blockIdx.x * blockDim.x + threadIdx.x;
  if (t < nAtoms * HID) {
    int i = t >> 7;
    atomicAdd(&sums[(size_t)bids[i] * HID + (t & 127)], x[t]);
  }
  if (t < nAtoms) atomicAdd(&cnt[bids[t]], 1.f);
}

__global__ void k_head(const float* __restrict__ sums, const float* __restrict__ cnt,
                       const float* __restrict__ Wp1, const float* __restrict__ bp1,
                       const float* __restrict__ Wp2, const float* __restrict__ bp2,
                       const float* __restrict__ Wp3, const float* __restrict__ bp3,
                       float* __restrict__ out, int nGraphs) {
  int g = threadIdx.x;
  if (g >= nGraphs) return;
  float pooled[HID], h1[64], h2[32];
  float c = fmaxf(cnt[g], 1.f);
  for (int h = 0; h < HID; ++h) pooled[h] = sums[g * HID + h] / c;
  for (int j = 0; j < 64; ++j) {
    float s = bp1[j];
    for (int h = 0; h < HID; ++h) s += pooled[h] * Wp1[h * 64 + j];
    h1[j] = fmaxf(s, 0.f);
  }
  for (int j = 0; j < 32; ++j) {
    float s = bp2[j];
    for (int h = 0; h < 64; ++h) s += h1[h] * Wp2[h * 32 + j];
    h2[j] = fmaxf(s, 0.f);
  }
  float s = bp3[0];
  for (int h = 0; h < 32; ++h) s += h2[h] * Wp3[h];
  out[g] = (s > 20.f) ? s : log1pf(expf(s));
}

// ---------------- host launcher ----------------

extern "C" void kernel_launch(void* const* d_in, const int* in_sizes, int n_in,
                              void* d_out, int out_size, void* d_ws, size_t ws_size,
                              hipStream_t stream) {
  const int N = in_sizes[0];           // atoms
  const int E = in_sizes[2];           // edges
  const int G = out_size;              // graphs

  const int*   atom_types = (const int*)d_in[0];
  const int*   edge_index = (const int*)d_in[1];
  const int*   e_src = edge_index;
  const int*   e_dst = edge_index + E;
  const float* dist  = (const float*)d_in[2];
  const int*   bids  = (const int*)d_in[3];
  const float* emb   = (const float*)d_in[4];
  // layer l param k: 5 + l*14 + k  (Wn,bn,We,be,Wm1,bm1,Wm2,bm2,Wu1,bu1,Wu2,bu2,ln_g,ln_b)
  const int LBASE = 5;
  const float* Wp1 = (const float*)d_in[LBASE + 4 * 14 + 0];
  const float* bp1 = (const float*)d_in[LBASE + 4 * 14 + 1];
  const float* Wp2 = (const float*)d_in[LBASE + 4 * 14 + 2];
  const float* bp2 = (const float*)d_in[LBASE + 4 * 14 + 3];
  const float* Wp3 = (const float*)d_in[LBASE + 4 * 14 + 4];
  const float* bp3 = (const float*)d_in[LBASE + 4 * 14 + 5];
  float* out = (float*)d_out;

  // workspace carve-out
  char* ws = (char*)d_ws;
  size_t off = 0;
  auto carve = [&](size_t bytes) -> void* {
    void* p = ws + off;
    off += (bytes + 255) & ~(size_t)255;
    return p;
  };
  const size_t NH = (size_t)N * HID;
  float*  xA     = (float*)carve(NH * 4);
  float*  xB     = (float*)carve(NH * 4);
  float*  agg    = (float*)carve(NH * 4);
  bf16_t* x_bf   = (bf16_t*)carve(NH * 2);
  bf16_t* xt_bf  = (bf16_t*)carve(NH * 2);
  bf16_t* agg_bf = (bf16_t*)carve(NH * 2);
  float*  sums   = (float*)carve((size_t)G * HID * 4);
  float*  cnt    = (float*)carve((size_t)G * 4);
  bf16_t* pW[4][5];  // Wn, Wm1, Wm2, Wu1, Wu2 packed per layer
  const int pK[5]   = {128, 256, 128, 256, 128};
  const int pOfs[5] = {0, 4, 6, 8, 10};
  for (int l = 0; l < 4; ++l)
    for (int m = 0; m < 5; ++m)
      pW[l][m] = (bf16_t*)carve((size_t)pK[m] * HID * 2);

  // ---- pack all layer weights to WMMA B-fragment layout (bf16)
  for (int l = 0; l < 4; ++l) {
    for (int m = 0; m < 5; ++m) {
      const float* Wsrc = (const float*)d_in[LBASE + l * 14 + pOfs[m]];
      int total = pK[m] * HID;
      k_pack<<<(total + 255) / 256, 256, 0, stream>>>(Wsrc, pW[l][m], pK[m]);
    }
  }

  // ---- embedding lookup
  k_embed<<<((int)NH + 255) / 256, 256, 0, stream>>>(atom_types, emb, xA, N);

  float* x_cur = xA;
  float* x_nxt = xB;
  const int nodeBlocks = (N + 127) / 128;
  const int edgeBlocks = (E + 127) / 128;

  for (int l = 0; l < 4; ++l) {
    const float* bn   = (const float*)d_in[LBASE + l * 14 + 1];
    const float* We   = (const float*)d_in[LBASE + l * 14 + 2];
    const float* be   = (const float*)d_in[LBASE + l * 14 + 3];
    const float* bm1  = (const float*)d_in[LBASE + l * 14 + 5];
    const float* bm2  = (const float*)d_in[LBASE + l * 14 + 7];
    const float* bu1  = (const float*)d_in[LBASE + l * 14 + 9];
    const float* bu2  = (const float*)d_in[LBASE + l * 14 + 11];
    const float* ln_g = (const float*)d_in[LBASE + l * 14 + 12];
    const float* ln_b = (const float*)d_in[LBASE + l * 14 + 13];

    // x -> bf16
    k_cast_bf16<<<((int)NH + 255) / 256, 256, 0, stream>>>(x_cur, x_bf, (int)NH);
    // xt = bf16(x @ Wn + bn)
    k_node_linear<<<nodeBlocks, TPB, 0, stream>>>(x_bf, pW[l][0], bn, xt_bf, N);
    // agg = 0
    k_zerof<<<((int)NH + 255) / 256, 256, 0, stream>>>(agg, (int)NH);
    // edge messages + scatter
    k_edge_message<<<edgeBlocks, TPB, 0, stream>>>(
        xt_bf, pW[l][1], pW[l][2], bm1, bm2, We, be, e_src, e_dst, dist, agg, E);
    // agg -> bf16
    k_cast_bf16<<<((int)NH + 255) / 256, 256, 0, stream>>>(agg, agg_bf, (int)NH);
    // node update (+LN, residual for l>0)
    k_node_update<<<nodeBlocks, TPB, 0, stream>>>(
        agg_bf, x_bf, x_cur, pW[l][3], pW[l][4], bu1, bu2, ln_g, ln_b,
        x_nxt, N, l > 0 ? 1 : 0);
    float* t = x_cur; x_cur = x_nxt; x_nxt = t;
  }

  // ---- pooling + head
  k_zerof<<<(G * HID + 255) / 256, 256, 0, stream>>>(sums, G * HID);
  k_zerof<<<(G + 255) / 256, 256, 0, stream>>>(cnt, G);
  k_pool<<<((int)NH + 255) / 256, 256, 0, stream>>>(x_cur, bids, sums, cnt, N);
  k_head<<<1, 64, 0, stream>>>(sums, cnt, Wp1, bp1, Wp2, bp2, Wp3, bp3, out, G);
}
